// HybridMoE_55542517071981
// MI455X (gfx1250) — matrix operand order repaired and checked by reading, not compiled, not run on hardware
//
#include <hip/hip_runtime.h>
#include <hip/hip_bf16.h>

// ---------------------------------------------------------------------------
// HybridMoE on gfx1250 (MI455X): route -> gather -> top-2-only expert MLP with
// v_wmma_f32_16x16x32_bf16. Weights pre-converted to bf16 and transposed
// (K-contiguous per output column) so WMMA B-fragments are two contiguous
// global_load_b128's; each B-fragment is reused across 2 M-subtiles (MT=32).
// ---------------------------------------------------------------------------

#define NUM_EXPERTS 8
#define NUM_LANGS   4
#define D_MODEL     1024
#define HID         4096
#define NTOK        4096        // B*S = 2*2048
#define MT          32          // tokens per tile (2 M-subtiles of 16)
#define SUB         2           // M-subtiles per tile
#define HC          128         // hidden chunk per iteration
#define XS_STRIDE   1032        // 1024 + 8 halfs pad (conflict-free LDS)
#define HS_STRIDE   136         // 128 + 8 halfs pad

typedef __attribute__((ext_vector_type(16))) __bf16 v16bf;
typedef __attribute__((ext_vector_type(8)))  float  v8f;

union Frag {
    unsigned short u[16];
    uint4          q[2];
    v16bf          v;
};

__device__ __forceinline__ unsigned short f2bf(float f) {
    unsigned u = __float_as_uint(f);
    u += 0x7FFFu + ((u >> 16) & 1u);     // round-to-nearest-even
    return (unsigned short)(u >> 16);
}

// ---------------------------------------------------------------------------
// Stage 0 (fast path): transpose + convert one expert matrix: in[e][R][C] f32
// -> out[e][C][R] bf16. 32x32 tiles via padded LDS; both sides coalesced.
// ---------------------------------------------------------------------------
__global__ __launch_bounds__(256) void transpose_bf16(
    const float* __restrict__ in, unsigned short* __restrict__ out, int R, int C)
{
    __shared__ unsigned short t[32][34];
    const size_t base = (size_t)blockIdx.z * R * C;
    const int c0 = blockIdx.x * 32;
    const int r0 = blockIdx.y * 32;
    const int tx  = threadIdx.x & 31;
    const int ty0 = threadIdx.x >> 5;
#pragma unroll
    for (int i = 0; i < 4; ++i) {
        const int r = ty0 + i * 8;
        t[r][tx] = f2bf(in[base + (size_t)(r0 + r) * C + c0 + tx]);
    }
    __syncthreads();
#pragma unroll
    for (int i = 0; i < 4; ++i) {
        const int c = ty0 + i * 8;
        out[base + (size_t)(c0 + c) * R + r0 + tx] = t[tx][c];
    }
}

// ---------------------------------------------------------------------------
// Stage 1: routing. One wave32 per token: lang logits (4) + router logits (8),
// softmaxes, top-2, normalized gates; append to per-expert buckets.
// ---------------------------------------------------------------------------
__global__ __launch_bounds__(256) void moe_route(
    const float* __restrict__ x,
    const float* __restrict__ router_w, const float* __restrict__ router_b,
    const float* __restrict__ lang_w,   const float* __restrict__ lang_b,
    int* __restrict__ counts, int* __restrict__ toks, float* __restrict__ gts)
{
    const int lane  = threadIdx.x & 31;
    const int wv    = threadIdx.x >> 5;
    const int token = blockIdx.x * 8 + wv;
    if (token >= NTOK) return;
    const float* xr = x + (size_t)token * D_MODEL;

    float acc[12];
#pragma unroll
    for (int j = 0; j < 12; ++j) acc[j] = 0.f;

    for (int d = lane; d < D_MODEL; d += 32) {
        const float xv = xr[d];
#pragma unroll
        for (int j = 0; j < NUM_LANGS; ++j)   acc[j]     += xv * lang_w[j * D_MODEL + d];
#pragma unroll
        for (int e = 0; e < NUM_EXPERTS; ++e) acc[4 + e] += xv * router_w[e * D_MODEL + d];
    }
#pragma unroll
    for (int off = 16; off > 0; off >>= 1)
#pragma unroll
        for (int j = 0; j < 12; ++j)
            acc[j] += __shfl_xor(acc[j], off, 32);

    if (lane == 0) {
        float ll[NUM_LANGS], m = -1e30f;
#pragma unroll
        for (int j = 0; j < NUM_LANGS; ++j) { ll[j] = acc[j] + lang_b[j]; m = fmaxf(m, ll[j]); }
        float s = 0.f;
#pragma unroll
        for (int j = 0; j < NUM_LANGS; ++j) { ll[j] = __expf(ll[j] - m); s += ll[j]; }
        float lp[NUM_LANGS + 1];
        const float inv = 1.f / s;
#pragma unroll
        for (int j = 0; j < NUM_LANGS; ++j) lp[j] = ll[j] * inv;
        lp[NUM_LANGS] = 0.f;

        float rl[NUM_EXPERTS], m2 = -1e30f;
#pragma unroll
        for (int e = 0; e < NUM_EXPERTS; ++e) {
            const int id = e & (NUM_LANGS - 1);
            const float ew = lp[id] + lp[id + 1];
            rl[e] = acc[4 + e] + router_b[e] + 0.1f * ew;
            m2 = fmaxf(m2, rl[e]);
        }
#pragma unroll
        for (int e = 0; e < NUM_EXPERTS; ++e) rl[e] = __expf(rl[e] - m2);

        int i1 = 0;
#pragma unroll
        for (int e = 1; e < NUM_EXPERTS; ++e) if (rl[e] > rl[i1]) i1 = e;
        int i2 = (i1 == 0) ? 1 : 0;
#pragma unroll
        for (int e = 0; e < NUM_EXPERTS; ++e)
            if (e != i1 && rl[e] > rl[i2]) i2 = e;

        const float p1 = rl[i1], p2 = rl[i2];
        const float ni = 1.f / (p1 + p2);

        int s1 = atomicAdd(&counts[i1], 1);
        toks[i1 * NTOK + s1] = token;
        gts [i1 * NTOK + s1] = p1 * ni;
        int s2 = atomicAdd(&counts[i2], 1);
        toks[i2 * NTOK + s2] = token;
        gts [i2 * NTOK + s2] = p2 * ni;
    }
}

// ---------------------------------------------------------------------------
// Stage 2: fused expert MLP on gathered tokens.
// Block = 256 threads (8 waves), tile = 32 tokens of one expert.
//   phase A: h_chunk[32,128] = gelu(x[32,1024] @ w1[:,chunk] + b1)  (bf16->LDS)
//   phase B: out_acc[32,1024] += h_chunk @ w2[chunk,:]
// Every B-fragment (weights) is reused by SUB=2 M-subtiles.
// PRE=true: B-fragments from pre-transposed bf16 weights (2x b128 loads).
// PRE=false: fallback, convert f32 weights in flight.
// ---------------------------------------------------------------------------
template <bool PRE>
__global__ __launch_bounds__(256) void moe_mlp(
    const float* __restrict__ x,
    const float* __restrict__ w1, const float* __restrict__ b1,
    const float* __restrict__ w2, const float* __restrict__ b2,
    const unsigned short* __restrict__ w1t,   // [E][HID][D_MODEL] bf16
    const unsigned short* __restrict__ w2t,   // [E][D_MODEL][HID] bf16
    const int* __restrict__ counts, const int* __restrict__ toks,
    const float* __restrict__ gts, float* __restrict__ out)
{
    const int e    = blockIdx.y;
    const int cnt  = counts[e];
    const int m0   = blockIdx.x * MT;
    if (m0 >= cnt) return;

    __shared__ unsigned short xs[MT * XS_STRIDE];   // x tile, bf16  (~64.5 KB)
    __shared__ unsigned short hs[MT * HS_STRIDE];   // h chunk, bf16 (~8.5 KB)
    __shared__ int   tok_s[MT];
    __shared__ float gate_s[MT];

    const int tid  = threadIdx.x;
    const int lane = tid & 31;
    const int wv   = tid >> 5;
    const int hi   = lane >> 4;        // half-wave selector
    const int nl   = lane & 15;        // N within tile
    const int mrow = lane & 15;

    if (tid < MT) {
        const int p     = m0 + tid;
        const int valid = (p < cnt);
        tok_s[tid]  = valid ? toks[e * NTOK + p] : toks[e * NTOK + m0];
        gate_s[tid] = valid ? gts [e * NTOK + p] : 0.f;
    }
    __syncthreads();

    // load x tile -> LDS bf16 (float4 per thread-iteration, coalesced)
    for (int i = tid; i < MT * (D_MODEL / 4); i += 256) {
        const int r  = i >> 8;
        const int c4 = i & 255;
        const float4 f = ((const float4*)(x + (size_t)tok_s[r] * D_MODEL))[c4];
        unsigned short* dst = &xs[r * XS_STRIDE + c4 * 4];
        dst[0] = f2bf(f.x); dst[1] = f2bf(f.y);
        dst[2] = f2bf(f.z); dst[3] = f2bf(f.w);
    }
    __syncthreads();

    const float* W1 = w1 + (size_t)e * D_MODEL * HID;
    const float* B1 = b1 + (size_t)e * HID;
    const float* W2 = w2 + (size_t)e * HID * D_MODEL;
    const float* B2 = b2 + (size_t)e * D_MODEL;
    const unsigned short* W1T = w1t + (size_t)e * D_MODEL * HID;
    const unsigned short* W2T = w2t + (size_t)e * HID * D_MODEL;

    v8f acc[8 * SUB];
#pragma unroll
    for (int t = 0; t < 8 * SUB; ++t) acc[t] = (v8f){};

    for (int it = 0; it < HID / HC; ++it) {
        // ---- phase A: one 32x16 h-tile per wave, K = D_MODEL ----
        const int ncol = it * HC + wv * 16 + nl;   // global hidden column
        v8f c[SUB];
#pragma unroll
        for (int s = 0; s < SUB; ++s) c[s] = (v8f){};
        for (int k0 = 0; k0 < D_MODEL; k0 += 32) {
            Frag b;
            if (PRE) {
                // w1t row = hidden column; 16 consecutive K -> 32 contiguous B
                const unsigned short* p = W1T + (size_t)ncol * D_MODEL + k0 + 16 * hi;
                b.q[0] = *(const uint4*)p;
                b.q[1] = *(const uint4*)(p + 8);
            } else {
                const int kb = k0 + 16 * hi;
#pragma unroll
                for (int j = 0; j < 16; ++j)
                    b.u[j] = f2bf(W1[(size_t)(kb + j) * HID + ncol]);
            }
#pragma unroll
            for (int s = 0; s < SUB; ++s) {
                Frag a;
                const int abase = (mrow + 16 * s) * XS_STRIDE + k0 + 16 * hi;
                a.q[0] = *(const uint4*)&xs[abase];
                a.q[1] = *(const uint4*)&xs[abase + 8];
                c[s] = __builtin_amdgcn_wmma_f32_16x16x32_bf16(
                           false, a.v, false, b.v, (short)0, c[s], false, false);
            }
        }
        const float bia = B1[ncol];
#pragma unroll
        for (int s = 0; s < SUB; ++s)
#pragma unroll
            for (int r = 0; r < 8; ++r) {
                const float v = c[s][r] + bia;
                const float g = 0.5f * v * (1.0f + erff(v * 0.70710678118f)); // exact GELU
                hs[(s * 16 + r + 8 * hi) * HS_STRIDE + wv * 16 + nl] = f2bf(g);
            }
        __syncthreads();

        // ---- phase B: wave owns 128 output columns; K = HC chunk ----
#pragma unroll
        for (int t = 0; t < 8; ++t) {
            const int n = wv * 128 + t * 16 + nl;
#pragma unroll
            for (int k0 = 0; k0 < HC; k0 += 32) {
                Frag b;
                if (PRE) {
                    const unsigned short* p =
                        W2T + (size_t)n * HID + it * HC + k0 + 16 * hi;
                    b.q[0] = *(const uint4*)p;
                    b.q[1] = *(const uint4*)(p + 8);
                } else {
                    const int kb = it * HC + k0 + 16 * hi;
#pragma unroll
                    for (int j = 0; j < 16; ++j)
                        b.u[j] = f2bf(W2[(size_t)(kb + j) * D_MODEL + n]);
                }
#pragma unroll
                for (int s = 0; s < SUB; ++s) {
                    Frag a;
                    const int abase = (mrow + 16 * s) * HS_STRIDE + k0 + 16 * hi;
                    a.q[0] = *(const uint4*)&hs[abase];
                    a.q[1] = *(const uint4*)&hs[abase + 8];
                    acc[t * SUB + s] = __builtin_amdgcn_wmma_f32_16x16x32_bf16(
                        false, a.v, false, b.v, (short)0, acc[t * SUB + s], false, false);
                }
            }
        }
        __syncthreads();
    }

    // ---- epilogue: gate * (acc + b2) -> atomicAdd (token in 2 expert lists) ----
#pragma unroll
    for (int t = 0; t < 8; ++t) {
        const int n = wv * 128 + t * 16 + nl;
        const float bia = B2[n];
#pragma unroll
        for (int s = 0; s < SUB; ++s)
#pragma unroll
            for (int r = 0; r < 8; ++r) {
                const int row = s * 16 + r + 8 * hi;
                if (m0 + row < cnt) {
                    const float val = gate_s[row] * (acc[t * SUB + s][r] + bia);
                    atomicAdd(&out[(size_t)tok_s[row] * D_MODEL + n], val);
                }
            }
    }
}

// ---------------------------------------------------------------------------
extern "C" void kernel_launch(void* const* d_in, const int* in_sizes, int n_in,
                              void* d_out, int out_size, void* d_ws, size_t ws_size,
                              hipStream_t stream) {
    const float* x        = (const float*)d_in[0];
    const float* router_w = (const float*)d_in[1];
    const float* router_b = (const float*)d_in[2];
    const float* lang_w   = (const float*)d_in[3];
    const float* lang_b   = (const float*)d_in[4];
    const float* w1       = (const float*)d_in[5];
    const float* b1       = (const float*)d_in[6];
    const float* w2       = (const float*)d_in[7];
    const float* b2       = (const float*)d_in[8];
    float* out = (float*)d_out;

    // workspace layout:
    //   counts[8]            @ 0
    //   toks[8][4096] (int)  @ 128
    //   gts [8][4096] (f32)  @ 128 + 131072
    //   w1t bf16 [8][4096][1024] @ W1T_OFF (aligned)
    //   w2t bf16 [8][1024][4096] @ W1T_OFF + WT_SZ
    const size_t W1T_OFF = 266240;                                  // 4K-aligned
    const size_t WT_SZ   = (size_t)NUM_EXPERTS * D_MODEL * HID * 2; // 64 MiB
    int*   counts = (int*)d_ws;
    int*   toks   = (int*)((char*)d_ws + 128);
    float* gts    = (float*)((char*)d_ws + 128 + (size_t)NUM_EXPERTS * NTOK * 4);
    unsigned short* w1t = (unsigned short*)((char*)d_ws + W1T_OFF);
    unsigned short* w2t = (unsigned short*)((char*)d_ws + W1T_OFF + WT_SZ);
    const bool pre = ws_size >= W1T_OFF + 2 * WT_SZ;

    hipMemsetAsync(d_ws, 0, 128, stream);
    hipMemsetAsync(d_out, 0, (size_t)out_size * sizeof(float), stream);

    // routing: 4096 tokens, 1 wave32 per token, 8 waves per block
    moe_route<<<NTOK / 8, 256, 0, stream>>>(x, router_w, router_b, lang_w, lang_b,
                                            counts, toks, gts);

    dim3 grid(NTOK / MT, NUM_EXPERTS);
    if (pre) {
        // one-time weight convert+transpose (bf16, K-contiguous per column)
        transpose_bf16<<<dim3(HID / 32, D_MODEL / 32, NUM_EXPERTS), 256, 0, stream>>>(
            w1, w1t, D_MODEL, HID);
        transpose_bf16<<<dim3(D_MODEL / 32, HID / 32, NUM_EXPERTS), 256, 0, stream>>>(
            w2, w2t, HID, D_MODEL);
        moe_mlp<true><<<grid, 256, 0, stream>>>(x, w1, b1, w2, b2, w1t, w2t,
                                                counts, toks, gts, out);
    } else {
        moe_mlp<false><<<grid, 256, 0, stream>>>(x, w1, b1, w2, b2, w1t, w2t,
                                                 counts, toks, gts, out);
    }
}
